// GLANT_24146306138835
// MI455X (gfx1250) — compile-verified, matrix operand order checked by reading, not run on hardware
//
#include <hip/hip_runtime.h>
#include <hip/hip_bf16.h>

#define N_NODES 50000
#define N_EDGES 400000
#define DIM_IN  128
#define HEADS   2
#define C1      128
#define C2      64
#define KHOPS   2

typedef __attribute__((ext_vector_type(16))) __bf16 v16bf;
typedef __attribute__((ext_vector_type(8)))  float  v8f;

// ---------- helpers ----------
__device__ __forceinline__ __bf16 f2bf(float f) {
    union { float f; unsigned u; } v; v.f = f;
    unsigned r = (v.u + 0x7FFFu + ((v.u >> 16) & 1u)) >> 16;
    union { unsigned short s; __bf16 b; } o; o.s = (unsigned short)r;
    return o.b;
}
__device__ __forceinline__ unsigned f2ord(float f) {
    unsigned u = __float_as_uint(f);
    return (u & 0x80000000u) ? ~u : (u | 0x80000000u);
}
__device__ __forceinline__ float ord2f(unsigned u) {
    unsigned v = (u & 0x80000000u) ? (u & 0x7FFFFFFFu) : ~u;
    return __uint_as_float(v);
}
__device__ __forceinline__ float wave_sum(float v) {
#pragma unroll
    for (int off = 16; off; off >>= 1) v += __shfl_xor(v, off, 32);
    return v;
}

// ---------- utility kernels ----------
__global__ void fill_u32_kernel(unsigned* p, unsigned val, long long n) {
    long long i = (long long)blockIdx.x * blockDim.x + threadIdx.x;
    if (i < n) p[i] = val;
}

__global__ void cvt_bf16_kernel(const float* __restrict__ in, __bf16* __restrict__ out, long long n) {
    long long i = (long long)blockIdx.x * blockDim.x + threadIdx.x;
    if (i < n) out[i] = f2bf(in[i]);
}

// Pack W [K, Ncols] f32 row-major -> WMMA B fragments [K/32][Ncols/16][32][16] bf16
__global__ void pack_w_kernel(const float* __restrict__ W, __bf16* __restrict__ Bp, int K, int Ncols) {
    int total = (K / 32) * (Ncols / 16) * 32 * 16;
    int i = blockIdx.x * blockDim.x + threadIdx.x;
    if (i >= total) return;
    int e    = i & 15;
    int lane = (i >> 4) & 31;
    int t    = i >> 9;
    int ntiles = Ncols / 16;
    int kt = t / ntiles, nt = t % ntiles;
    int h = lane >> 4;
    int n = nt * 16 + (lane & 15);
    int k = kt * 32 + ((e >> 3) << 4) + 8 * h + (e & 7);
    Bp[i] = f2bf(W[(size_t)k * Ncols + n]);
}

// ---------- WMMA GEMM, 1x4 register-blocked ----------
// C = A(bf16,[M,K]) @ Bp(packed bf16) + bias ; C f32 [M,Ncols]; Ncols % 64 == 0
__global__ void wmma_gemm_bias_kernel(const __bf16* __restrict__ A, const __bf16* __restrict__ Bp,
                                      const float* __restrict__ bias, float* __restrict__ C,
                                      int M, int K, int Ncols) {
    int wid  = blockIdx.x * (blockDim.x >> 5) + (threadIdx.x >> 5);
    int lane = threadIdx.x & 31;
    int ntiles  = Ncols / 16;
    int ngroups = ntiles / 4;                 // 4 N-tiles per wave
    int totalWork = (M / 16) * ngroups;
    if (wid >= totalWork) return;             // uniform per-wave: EXEC stays all-ones
    int tileM = wid / ngroups;
    int group = wid % ngroups;
    int h = lane >> 4;

    v8f acc0 = {}, acc1 = {}, acc2 = {}, acc3 = {};
    const __bf16* arow  = A + (size_t)(tileM * 16 + (lane & 15)) * K;
    const __bf16* bbase = Bp + (size_t)(group * 4) * 512 + (size_t)lane * 16;
    size_t bstride = (size_t)ntiles * 512;

    for (int kt = 0; kt < K / 32; ++kt) {
        const __bf16* ap = arow + kt * 32 + 8 * h;
        v16bf a;
#pragma unroll
        for (int e = 0; e < 8; ++e) { a[e] = ap[e]; a[8 + e] = ap[16 + e]; }
        const __bf16* bp = bbase + (size_t)kt * bstride;
        v16bf b0 = *(const v16bf*)(bp);
        v16bf b1 = *(const v16bf*)(bp + 512);
        v16bf b2 = *(const v16bf*)(bp + 1024);
        v16bf b3 = *(const v16bf*)(bp + 1536);
        acc0 = __builtin_amdgcn_wmma_f32_16x16x32_bf16(false, a, false, b0, (short)0, acc0, false, false);
        acc1 = __builtin_amdgcn_wmma_f32_16x16x32_bf16(false, a, false, b1, (short)0, acc1, false, false);
        acc2 = __builtin_amdgcn_wmma_f32_16x16x32_bf16(false, a, false, b2, (short)0, acc2, false, false);
        acc3 = __builtin_amdgcn_wmma_f32_16x16x32_bf16(false, a, false, b3, (short)0, acc3, false, false);
    }

#define STORE_TILE(ACC, T)                                                   \
    {                                                                        \
        int n = (group * 4 + (T)) * 16 + (lane & 15);                        \
        float bn = bias[n];                                                  \
        _Pragma("unroll")                                                    \
        for (int r = 0; r < 8; ++r) {                                        \
            int m = tileM * 16 + r + 8 * h;                                  \
            C[(size_t)m * Ncols + n] = ACC[r] + bn;                          \
        }                                                                    \
    }
    STORE_TILE(acc0, 0)
    STORE_TILE(acc1, 1)
    STORE_TILE(acc2, 2)
    STORE_TILE(acc3, 3)
#undef STORE_TILE
}

// ---------- edge phase ----------
// one wave per edge: e_h = sum_c att[h,c]*leaky(xl[s,h,c]+xr[d,h,c]); atomic-max into mord
__global__ void edge_logit_kernel(const float* __restrict__ xl, const float* __restrict__ xr,
                                  const int* __restrict__ src, const int* __restrict__ dst,
                                  int nE, int nReal, const float* __restrict__ att,
                                  float* __restrict__ elog, unsigned* __restrict__ mord, int C) {
    int wid  = blockIdx.x * (blockDim.x >> 5) + (threadIdx.x >> 5);
    int lane = threadIdx.x & 31;
    if (wid >= nE) return;
    int s, d;
    if (wid < nReal) { s = src[wid]; d = dst[wid]; } else { s = d = wid - nReal; }
    int HC = 2 * C;
    float a0 = 0.f, a1 = 0.f;
    for (int c = lane; c < C; c += 32) {
        float v0 = xl[(size_t)s * HC + c] + xr[(size_t)d * HC + c];
        v0 = v0 > 0.f ? v0 : 0.2f * v0;
        a0 += att[c] * v0;
        float v1 = xl[(size_t)s * HC + C + c] + xr[(size_t)d * HC + C + c];
        v1 = v1 > 0.f ? v1 : 0.2f * v1;
        a1 += att[C + c] * v1;
    }
    a0 = wave_sum(a0); a1 = wave_sum(a1);
    if (lane == 0) {
        elog[(size_t)wid * 2 + 0] = a0;
        elog[(size_t)wid * 2 + 1] = a1;
        atomicMax(&mord[(size_t)d * 2 + 0], f2ord(a0));
        atomicMax(&mord[(size_t)d * 2 + 1], f2ord(a1));
    }
}

__global__ void edge_exp_kernel(float* __restrict__ elog, const unsigned* __restrict__ mord,
                                const int* __restrict__ dst, int nE, int nReal,
                                float* __restrict__ z) {
    int i = blockIdx.x * blockDim.x + threadIdx.x;
    if (i >= nE * 2) return;
    int e = i >> 1, h = i & 1;
    int d = (e < nReal) ? dst[e] : e - nReal;
    float m = ord2f(mord[(size_t)d * 2 + h]);
    if (!__builtin_isfinite(m)) m = 0.f;
    float p = __expf(elog[i] - m);
    elog[i] = p;
    atomicAdd(&z[(size_t)d * 2 + h], p);
}

__global__ void edge_scatter_kernel(const float* __restrict__ xl, const float* __restrict__ elog,
                                    const float* __restrict__ z, const int* __restrict__ src,
                                    const int* __restrict__ dst, int nE, int nReal,
                                    float* __restrict__ acc, int C) {
    int wid  = blockIdx.x * (blockDim.x >> 5) + (threadIdx.x >> 5);
    int lane = threadIdx.x & 31;
    if (wid >= nE) return;
    int s, d;
    if (wid < nReal) { s = src[wid]; d = dst[wid]; } else { s = d = wid - nReal; }
    float al0 = elog[(size_t)wid * 2 + 0] / (z[(size_t)d * 2 + 0] + 1e-16f);
    float al1 = elog[(size_t)wid * 2 + 1] / (z[(size_t)d * 2 + 1] + 1e-16f);
    int HC = 2 * C;
    for (int c = lane; c < C; c += 32) {
        atomicAdd(&acc[(size_t)d * HC + c],     al0 * xl[(size_t)s * HC + c]);
        atomicAdd(&acc[(size_t)d * HC + C + c], al1 * xl[(size_t)s * HC + C + c]);
    }
}

// msgs[n,k,c] = mean_h acc[n,h,c] + bias[c]
__global__ void msgs_final_kernel(const float* __restrict__ acc, const float* __restrict__ bias,
                                  float* __restrict__ msgs, int nNodes, int C, int k) {
    int i = blockIdx.x * blockDim.x + threadIdx.x;
    if (i >= nNodes * C) return;
    int n = i / C, c = i % C;
    float v = 0.5f * (acc[(size_t)n * 2 * C + c] + acc[(size_t)n * 2 * C + C + c]) + bias[c];
    msgs[((size_t)n * KHOPS + k) * C + c] = v;
}

// hop-gate softmax: w[n,k], K=2
__global__ void gate_kernel(const float* __restrict__ x, const float* __restrict__ Wg,
                            const float* __restrict__ bg, float* __restrict__ w,
                            int nNodes, int D) {
    int wid  = blockIdx.x * (blockDim.x >> 5) + (threadIdx.x >> 5);
    int lane = threadIdx.x & 31;
    if (wid >= nNodes) return;
    float g0 = 0.f, g1 = 0.f;
    for (int c = lane; c < D; c += 32) {
        float xv = x[(size_t)wid * D + c];
        g0 += xv * Wg[(size_t)c * 2 + 0];
        g1 += xv * Wg[(size_t)c * 2 + 1];
    }
    g0 = wave_sum(g0); g1 = wave_sum(g1);
    if (lane == 0) {
        g0 += bg[0]; g1 += bg[1];
        float mx = fmaxf(g0, g1);
        float e0 = __expf(g0 - mx), e1 = __expf(g1 - mx);
        float zz = e0 + e1;
        w[(size_t)wid * 2 + 0] = e0 / zz;
        w[(size_t)wid * 2 + 1] = e1 / zz;
    }
}

__global__ void combine_kernel(const float* __restrict__ msgs, const float* __restrict__ w,
                               float* __restrict__ out, int nNodes, int C, int doRelu) {
    int i = blockIdx.x * blockDim.x + threadIdx.x;
    if (i >= nNodes * C) return;
    int n = i / C, c = i % C;
    float v = msgs[((size_t)n * KHOPS + 0) * C + c] * w[(size_t)n * 2 + 0]
            + msgs[((size_t)n * KHOPS + 1) * C + c] * w[(size_t)n * 2 + 1];
    if (doRelu) v = fmaxf(v, 0.f);
    out[(size_t)n * C + c] = v;
}

// ---------- launch ----------
static inline int cdiv(long long a, long long b) { return (int)((a + b - 1) / b); }

extern "C" void kernel_launch(void* const* d_in, const int* in_sizes, int n_in,
                              void* d_out, int out_size, void* d_ws, size_t ws_size,
                              hipStream_t stream) {
    const float* x      = (const float*)d_in[0];
    const int*   edge0  = (const int*)d_in[1];   // [2,E]
    const int*   edge1  = (const int*)d_in[2];
    const float* Wl1    = (const float*)d_in[3];
    const float* bl1    = (const float*)d_in[4];
    const float* Wr1    = (const float*)d_in[5];
    const float* br1    = (const float*)d_in[6];
    const float* att1[2]  = { (const float*)d_in[7],  (const float*)d_in[8]  };
    const float* bias1[2] = { (const float*)d_in[9],  (const float*)d_in[10] };
    const float* Wg1    = (const float*)d_in[11];
    const float* bg1    = (const float*)d_in[12];
    const float* Wl2    = (const float*)d_in[13];
    const float* bl2    = (const float*)d_in[14];
    const float* Wr2    = (const float*)d_in[15];
    const float* br2    = (const float*)d_in[16];
    const float* att2[2]  = { (const float*)d_in[17], (const float*)d_in[18] };
    const float* bias2[2] = { (const float*)d_in[19], (const float*)d_in[20] };
    const float* Wg2    = (const float*)d_in[21];
    const float* bg2    = (const float*)d_in[22];
    float* out = (float*)d_out;

    // ---- workspace carve ----
    unsigned char* p = (unsigned char*)d_ws;
    auto carve = [&](size_t bytes) -> void* {
        void* r = (void*)p;
        p += (bytes + 255) & ~(size_t)255;
        return r;
    };
    __bf16* xbf   = (__bf16*)carve((size_t)N_NODES * DIM_IN * 2);        // also reused for h-bf16
    __bf16* WlP1  = (__bf16*)carve((size_t)4 * 16 * 512 * 2);
    __bf16* WrP1  = (__bf16*)carve((size_t)4 * 16 * 512 * 2);
    __bf16* WlP2  = (__bf16*)carve((size_t)4 * 8 * 512 * 2);
    __bf16* WrP2  = (__bf16*)carve((size_t)4 * 8 * 512 * 2);
    float*  xl    = (float*)carve((size_t)N_NODES * HEADS * C1 * 4);
    float*  xr    = (float*)carve((size_t)N_NODES * HEADS * C1 * 4);
    float*  elog  = (float*)carve((size_t)(N_EDGES + N_NODES) * 2 * 4);
    unsigned* mord = (unsigned*)carve((size_t)N_NODES * 2 * 4);
    float*  zbuf  = (float*)carve((size_t)N_NODES * 2 * 4);
    float*  accb  = (float*)carve((size_t)N_NODES * HEADS * C1 * 4);
    float*  msgs  = (float*)carve((size_t)N_NODES * KHOPS * C1 * 4);
    float*  gatew = (float*)carve((size_t)N_NODES * 2 * 4);
    float*  hbuf  = (float*)carve((size_t)N_NODES * C1 * 4);

    const int TB = 256;
    const unsigned ORD_NEG_INF = 0x007FFFFFu;  // f2ord(-inf)

    // ---- weight packing (once per launch; deterministic) ----
    pack_w_kernel<<<cdiv(4 * 16 * 512, TB), TB, 0, stream>>>(Wl1, WlP1, DIM_IN, HEADS * C1);
    pack_w_kernel<<<cdiv(4 * 16 * 512, TB), TB, 0, stream>>>(Wr1, WrP1, DIM_IN, HEADS * C1);
    pack_w_kernel<<<cdiv(4 * 8 * 512, TB), TB, 0, stream>>>(Wl2, WlP2, C1, HEADS * C2);
    pack_w_kernel<<<cdiv(4 * 8 * 512, TB), TB, 0, stream>>>(Wr2, WrP2, C1, HEADS * C2);

    auto run_layer = [&](const float* xin_f32, const __bf16* xin_bf,
                         const __bf16* WlP, const float* bl,
                         const __bf16* WrP, const float* br,
                         const float* const* att, const float* const* bias,
                         const float* Wg, const float* bg,
                         int C, float* dst_out, int doRelu,
                         const int* e0, const int* e1) {
        int HC = HEADS * C;
        // projections via WMMA (1x4 blocked: one wave per 16x64 output strip)
        int work = (N_NODES / 16) * (HC / 64);
        wmma_gemm_bias_kernel<<<cdiv(work, 8), TB, 0, stream>>>(xin_bf, WlP, bl, xl, N_NODES, 128, HC);
        wmma_gemm_bias_kernel<<<cdiv(work, 8), TB, 0, stream>>>(xin_bf, WrP, br, xr, N_NODES, 128, HC);
        // hop gate
        gate_kernel<<<cdiv(N_NODES, 8), TB, 0, stream>>>(xin_f32, Wg, bg, gatew, N_NODES, 128);

        for (int k = 0; k < KHOPS; ++k) {
            const int* src = (k == 0) ? e0 : e1;
            const int* dst = src + N_EDGES;
            int nE = (k == 0) ? (N_EDGES + N_NODES) : N_EDGES;  // hop0 has self loops
            fill_u32_kernel<<<cdiv((long long)N_NODES * 2, TB), TB, 0, stream>>>(mord, ORD_NEG_INF, (long long)N_NODES * 2);
            fill_u32_kernel<<<cdiv((long long)N_NODES * 2, TB), TB, 0, stream>>>((unsigned*)zbuf, 0u, (long long)N_NODES * 2);
            fill_u32_kernel<<<cdiv((long long)N_NODES * HC, TB), TB, 0, stream>>>((unsigned*)accb, 0u, (long long)N_NODES * HC);

            edge_logit_kernel<<<cdiv(nE, 8), TB, 0, stream>>>(xl, xr, src, dst, nE, N_EDGES, att[k], elog, mord, C);
            edge_exp_kernel<<<cdiv((long long)nE * 2, TB), TB, 0, stream>>>(elog, mord, dst, nE, N_EDGES, zbuf);
            edge_scatter_kernel<<<cdiv(nE, 8), TB, 0, stream>>>(xl, elog, zbuf, src, dst, nE, N_EDGES, accb, C);
            msgs_final_kernel<<<cdiv((long long)N_NODES * C, TB), TB, 0, stream>>>(accb, bias[k], msgs, N_NODES, C, k);
        }
        combine_kernel<<<cdiv((long long)N_NODES * C, TB), TB, 0, stream>>>(msgs, gatew, dst_out, N_NODES, C, doRelu);
    };

    // ---- layer 1 ----
    cvt_bf16_kernel<<<cdiv((long long)N_NODES * DIM_IN, TB), TB, 0, stream>>>(x, xbf, (long long)N_NODES * DIM_IN);
    run_layer(x, xbf, WlP1, bl1, WrP1, br1, att1, bias1, Wg1, bg1, C1, hbuf, /*relu=*/1, edge0, edge1);

    // ---- layer 2 ----
    cvt_bf16_kernel<<<cdiv((long long)N_NODES * C1, TB), TB, 0, stream>>>(hbuf, xbf, (long long)N_NODES * C1);
    run_layer(hbuf, xbf, WlP2, bl2, WrP2, br2, att2, bias2, Wg2, bg2, C2, out, /*relu=*/0, edge0, edge1);
}
